// RFc_14353780703409
// MI455X (gfx1250) — compile-verified, matrix operand order checked by reading, not run on hardware
//
#include <hip/hip_runtime.h>
#include <cstdint>
#include <cstddef>

typedef __bf16 bf16;
typedef __attribute__((ext_vector_type(16))) __bf16 v16bf;
typedef __attribute__((ext_vector_type(8)))  __bf16 v8bf;
typedef __attribute__((ext_vector_type(4)))  __bf16 v4bf;
typedef __attribute__((ext_vector_type(8)))  float   v8f;

#if __has_builtin(__builtin_amdgcn_tensor_load_to_lds) && __has_builtin(__builtin_amdgcn_s_wait_tensorcnt)
#define HAVE_TDM 1
typedef unsigned int v4u __attribute__((ext_vector_type(4)));
typedef int          v8i __attribute__((ext_vector_type(8)));
typedef int          v4i __attribute__((ext_vector_type(4)));
#else
#define HAVE_TDM 0
#endif

#define TB 256

// ---------------------------------------------------------------------------
// Implicit-GEMM conv3x3 (SAME, stride1) over NHWC bf16 activations.
// A rows = output pixels of BOTH streams concatenated: M = 2*N*H*W.
// K = 9*Cin (tap-major, channel contiguous). Weights pre-laid as [Cout][K].
// Bias (A_p) added only to rows < N*H*W (the y1 stream).
// Tile: BM=128 x BN=64 x BK=32. 8 waves (4x2); each wave owns a 32x32 C tile
// -> 2x2 register tiling, 4 v_wmma per K-step with A/B fragment reuse.
// LDS rows padded to 40 bf16 (80B): 16B-aligned b128 reads, conflict-free.
// B tile is DMA'd by the Tensor Data Mover (wave 0 issues tensor_load_to_lds,
// D# pad fields reproduce the 64B-data+16B-pad LDS row layout), synced with
// s_wait_tensorcnt + workgroup barrier.
// ---------------------------------------------------------------------------
__global__ __launch_bounds__(TB)
void conv_wmma(const bf16* __restrict__ act, const bf16* __restrict__ wnk,
               const float* __restrict__ bias, float* __restrict__ out,
               int N, int H, int W, int Cin, int Cout)
{
    const int HW  = H * W;
    const int NHW = N * HW;
    const int M   = 2 * NHW;
    const int K   = 9 * Cin;
    const int m0  = blockIdx.x * 128;
    const int n0  = blockIdx.y * 64;

    __shared__ __align__(16) bf16 As[128][40];
    __shared__ __align__(16) bf16 Bs[64][40];

    const int t    = threadIdx.x;
    const int lane = t & 31;
    const int wv   = t >> 5;
    const int wm   = (wv & 3) * 32;     // wave row offset (2 sub-tiles)
    const int wn   = (wv >> 2) * 32;    // wave col offset (2 sub-tiles)
    const int half = lane >> 4;
    const int l15  = lane & 15;

    v8f acc[2][2] = {};

    const bool fast = (Cin & 7) == 0;   // 8-wide K chunk stays inside one tap

    // A-tile loader: 256 threads x two 8-elem chunks = 128x32 tile
    const int a_m  = t >> 1;            // 0..127
    const int a_kc = (t & 1) * 16;      // 0 or 16
    const int grow = m0 + a_m;
    const bool rowok = grow < M;
    int sOfs = 0, pn = 0, ph = 0, pw = 0;
    if (rowok) {
        int r = grow;
        if (r >= NHW) { r -= NHW; sOfs = NHW; }
        pn = r / HW; int hw = r - pn * HW;
        ph = hw / W; pw = hw - ph * W;
    }

#if !HAVE_TDM
    // Fallback B loader: 256 threads x one 8-elem chunk = 64x32 tile
    const int b_n  = t >> 2;            // 0..63
    const int b_kc = (t & 3) * 8;
    const int gn   = n0 + b_n;
#endif

    for (int kk = 0; kk < K; kk += 32) {
        // ---- stage A (im2col gather) ----
#pragma unroll
        for (int cc = 0; cc < 2; cc++) {
            v8bf av = {};
            const int kc = a_kc + cc * 8;
            if (rowok) {
                const int k = kk + kc;
                if (k < K) {
                    if (fast) {
                        int tap = k / Cin;
                        int ci  = k - tap * Cin;
                        int dy  = tap / 3, dx = tap - dy * 3;
                        int ih  = ph + dy - 1, iw = pw + dx - 1;
                        if ((unsigned)ih < (unsigned)H && (unsigned)iw < (unsigned)W) {
                            const bf16* src =
                                act + ((size_t)(sOfs + (pn * H + ih) * W + iw)) * Cin + ci;
                            av = *(const v8bf*)src;
                            __builtin_prefetch(src + 32, 0, 1);   // next K-step
                        }
                    } else {
#pragma unroll
                        for (int e = 0; e < 8; e++) {
                            const int ke = k + e;
                            float v = 0.f;
                            if (ke < K) {
                                int tap = ke / Cin;
                                int ci  = ke - tap * Cin;
                                int dy  = tap / 3, dx = tap - dy * 3;
                                int ih  = ph + dy - 1, iw = pw + dx - 1;
                                if ((unsigned)ih < (unsigned)H && (unsigned)iw < (unsigned)W)
                                    v = (float)act[((size_t)(sOfs + (pn * H + ih) * W + iw)) * Cin + ci];
                            }
                            av[e] = (bf16)v;
                        }
                    }
                }
            }
            *(v8bf*)&As[a_m][kc] = av;
        }

        // ---- stage B: Tensor Data Mover DMA of the 64x32 weight tile ----
#if HAVE_TDM
        if (t < 32) {
            const unsigned long long gaddr =
                (unsigned long long)(uintptr_t)(wnk + (size_t)n0 * K + kk);
            const unsigned int ldsa = (unsigned int)(uintptr_t)&Bs[0][0];
            int td0 = K - kk;                       // dim0 OOB => zero-fill tail

            v4u g0;
            g0[0] = 1u;                             // count=1, user mode
            g0[1] = ldsa;                           // lds_addr
            g0[2] = (unsigned int)gaddr;            // global_addr[31:0]
            g0[3] = (unsigned int)(gaddr >> 32) | (2u << 30);   // addr msbs | type=2

            v8i g1;
            g1[0] = (1 << 16)        // data_size = 2 bytes
                  | (1 << 20)        // pad_enable
                  | (3 << 22)        // pad_interval: 16 DWORDs (64B of data)
                  | (3 << 25);       // pad_amount:   4 DWORDs (16B pad)
            g1[1] = (td0 & 0xffff) << 16;                 // tensor_dim0[15:0]
            g1[2] = ((td0 >> 16) & 0xffff) | (64 << 16);  // dim0 msbs | tensor_dim1=64
            g1[3] = (32 << 16);                           // tile_dim0 = 32
            g1[4] = 64;                                   // tile_dim1 = 64 rows
            g1[5] = K;                                    // tensor_dim0_stride
            g1[6] = 0;
            g1[7] = 0;

            v4i z4 = {};
            v8i z8 = {};
            __builtin_amdgcn_tensor_load_to_lds(g0, g1, z4, z4, z8, 0);
            __builtin_amdgcn_s_wait_tensorcnt((short)0);
        }
#else
        {
            v8bf bv = {};
            if (gn < Cout) {
                if ((K & 7) == 0) {
                    if (kk + b_kc < K)
                        bv = *(const v8bf*)(wnk + (size_t)gn * K + kk + b_kc);
                } else {
#pragma unroll
                    for (int e = 0; e < 8; e++) {
                        const int ke = kk + b_kc + e;
                        bv[e] = (ke < K) ? wnk[(size_t)gn * K + ke] : (bf16)0.f;
                    }
                }
            }
            *(v8bf*)&Bs[b_n][b_kc] = bv;
        }
#endif

        __syncthreads();

        // ---- fragments (documented 16-bit A/B VGPR layouts) + 4x WMMA ----
        v16bf af[2], bfr[2];
#pragma unroll
        for (int i = 0; i < 2; i++) {
            v8bf lo = *(const v8bf*)&As[wm + 16 * i + l15][half * 8];
            v8bf hi = *(const v8bf*)&As[wm + 16 * i + l15][16 + half * 8];
#pragma unroll
            for (int e = 0; e < 8; e++) { af[i][e] = lo[e]; af[i][e + 8] = hi[e]; }
        }
#pragma unroll
        for (int j = 0; j < 2; j++) {
            v8bf lo = *(const v8bf*)&Bs[wn + 16 * j + l15][half * 8];
            v8bf hi = *(const v8bf*)&Bs[wn + 16 * j + l15][16 + half * 8];
#pragma unroll
            for (int e = 0; e < 8; e++) { bfr[j][e] = lo[e]; bfr[j][e + 8] = hi[e]; }
        }
#pragma unroll
        for (int i = 0; i < 2; i++)
#pragma unroll
            for (int j = 0; j < 2; j++)
                acc[i][j] = __builtin_amdgcn_wmma_f32_16x16x32_bf16(
                    false, af[i], false, bfr[j], (short)0, acc[i][j], false, false);

        __syncthreads();
    }

    // ---- store C (f32), add y1 bias on first half of M ----
#pragma unroll
    for (int i = 0; i < 2; i++) {
#pragma unroll
        for (int j = 0; j < 2; j++) {
            const int cn = n0 + wn + 16 * j + l15;
#pragma unroll
            for (int rI = 0; rI < 8; rI++) {
                const int row = m0 + wm + 16 * i + half * 8 + rI;
                if (row < M && cn < Cout) {
                    float v = acc[i][j][rI];
                    if (row < NHW) v += bias[cn];
                    out[(size_t)row * Cout + cn] = v;
                }
            }
        }
    }
}

// ---------------------------------------------------------------------------
// FC GEMM: A bf16 [M][K], W bf16 [Nn][K], out f32 [M][Nn], bias on rows < Mb.
// Same 128x64 tiling, manual loaders (K % 32 == 0 for all FC layers).
// ---------------------------------------------------------------------------
__global__ __launch_bounds__(TB)
void fc_wmma(const bf16* __restrict__ A, const bf16* __restrict__ Wnk,
             const float* __restrict__ bias, float* __restrict__ out,
             int M, int Mb, int K, int Nn)
{
    const int m0 = blockIdx.x * 128;
    const int n0 = blockIdx.y * 64;

    __shared__ __align__(16) bf16 As[128][40];
    __shared__ __align__(16) bf16 Bs[64][40];

    const int t    = threadIdx.x;
    const int lane = t & 31;
    const int wv   = t >> 5;
    const int wm   = (wv & 3) * 32;
    const int wn   = (wv >> 2) * 32;
    const int half = lane >> 4;
    const int l15  = lane & 15;

    v8f acc[2][2] = {};

    const int a_m  = t >> 1;
    const int a_kc = (t & 1) * 16;
    const int arow = m0 + a_m;
    const int b_n  = t >> 2;
    const int b_kc = (t & 3) * 8;
    const int gn   = n0 + b_n;

    for (int kk = 0; kk < K; kk += 32) {
#pragma unroll
        for (int cc = 0; cc < 2; cc++) {
            v8bf av = {};
            const int k = kk + a_kc + cc * 8;
            if (arow < M && k < K)
                av = *(const v8bf*)(A + (size_t)arow * K + k);
            *(v8bf*)&As[a_m][a_kc + cc * 8] = av;
        }
        {
            v8bf bv = {};
            if (gn < Nn && kk + b_kc < K)
                bv = *(const v8bf*)(Wnk + (size_t)gn * K + kk + b_kc);
            *(v8bf*)&Bs[b_n][b_kc] = bv;
        }

        __syncthreads();

        v16bf af[2], bfr[2];
#pragma unroll
        for (int i = 0; i < 2; i++) {
            v8bf lo = *(const v8bf*)&As[wm + 16 * i + l15][half * 8];
            v8bf hi = *(const v8bf*)&As[wm + 16 * i + l15][16 + half * 8];
#pragma unroll
            for (int e = 0; e < 8; e++) { af[i][e] = lo[e]; af[i][e + 8] = hi[e]; }
        }
#pragma unroll
        for (int j = 0; j < 2; j++) {
            v8bf lo = *(const v8bf*)&Bs[wn + 16 * j + l15][half * 8];
            v8bf hi = *(const v8bf*)&Bs[wn + 16 * j + l15][16 + half * 8];
#pragma unroll
            for (int e = 0; e < 8; e++) { bfr[j][e] = lo[e]; bfr[j][e + 8] = hi[e]; }
        }
#pragma unroll
        for (int i = 0; i < 2; i++)
#pragma unroll
            for (int j = 0; j < 2; j++)
                acc[i][j] = __builtin_amdgcn_wmma_f32_16x16x32_bf16(
                    false, af[i], false, bfr[j], (short)0, acc[i][j], false, false);

        __syncthreads();
    }

#pragma unroll
    for (int i = 0; i < 2; i++) {
#pragma unroll
        for (int j = 0; j < 2; j++) {
            const int cn = n0 + wn + 16 * j + l15;
#pragma unroll
            for (int rI = 0; rI < 8; rI++) {
                const int row = m0 + wm + 16 * i + half * 8 + rI;
                if (row < M && cn < Nn) {
                    float v = acc[i][j][rI];
                    if (row < Mb) v += bias[cn];
                    out[(size_t)row * Nn + cn] = v;
                }
            }
        }
    }
}

// ---------------------------------------------------------------------------
// BN batch stats per (stream, channel): biased mean/var over N*H*W rows.
// ---------------------------------------------------------------------------
__global__ void bn_stats(const float* __restrict__ y, int rows, int C,
                         float* __restrict__ mean, float* __restrict__ rstd)
{
    const int b = blockIdx.x;
    const int s = b / C, c = b - s * C;
    const float* base = y + (size_t)s * rows * C + c;
    float sum = 0.f, sq = 0.f;
    for (int r = threadIdx.x; r < rows; r += blockDim.x) {
        float v = base[(size_t)r * C];
        sum += v; sq += v * v;
    }
    __shared__ float sh1[256], sh2[256];
    sh1[threadIdx.x] = sum; sh2[threadIdx.x] = sq;
    __syncthreads();
    for (int o = 128; o > 0; o >>= 1) {
        if (threadIdx.x < o) {
            sh1[threadIdx.x] += sh1[threadIdx.x + o];
            sh2[threadIdx.x] += sh2[threadIdx.x + o];
        }
        __syncthreads();
    }
    if (threadIdx.x == 0) {
        float m = sh1[0] / (float)rows;
        float v = sh2[0] / (float)rows - m * m;
        mean[b] = m;
        rstd[b] = rsqrtf(v + 1e-5f);
    }
}

// BN apply + relu/sign coupling (non-pool layers). half = N*H*W*C elements.
__global__ void bn_act(const float* __restrict__ y, const float* __restrict__ mean,
                       const float* __restrict__ rstd, const float* __restrict__ g,
                       const float* __restrict__ bb, bf16* __restrict__ out,
                       size_t half, int C)
{
    for (size_t i = blockIdx.x * (size_t)blockDim.x + threadIdx.x; i < half;
         i += (size_t)gridDim.x * blockDim.x) {
        const int c = (int)(i % C);
        const float y1 = (y[i]        - mean[c])     * rstd[c]     * g[c] + bb[c];
        const float y2 = (y[i + half] - mean[C + c]) * rstd[C + c] * g[c] + bb[c];
        out[i]        = (bf16)(y1 > 0.f ? y1 : 0.f);
        out[i + half] = (bf16)(y1 > 0.f ? y2 : 0.f);
    }
}

// Fused BN + mmp/zfo/pool stage. mmp(p)==1 iff p is the STRICT max of its
// aligned 2x2 block; t = mmp(y1)*y2; x2 = sign(blocksum(sign t))*max|t|*[x1>0].
__global__ void bn_pool(const float* __restrict__ y, const float* __restrict__ mean,
                        const float* __restrict__ rstd, const float* __restrict__ g,
                        const float* __restrict__ bb, bf16* __restrict__ out,
                        int N, int H, int W, int C)
{
    const int Ho = H >> 1, Wo = W >> 1;
    const size_t tot    = (size_t)N * Ho * Wo * C;
    const size_t inHalf = (size_t)N * H * W * C;
    for (size_t i = blockIdx.x * (size_t)blockDim.x + threadIdx.x; i < tot;
         i += (size_t)gridDim.x * blockDim.x) {
        const int c = (int)(i % C);
        size_t r = i / C;
        const int wo = (int)(r % Wo); r /= Wo;
        const int ho = (int)(r % Ho);
        const int n  = (int)(r / Ho);
        const float m1 = mean[c], s1 = rstd[c], m2 = mean[C + c], s2 = rstd[C + c];
        const float gg = g[c], bc = bb[c];
        float a1[4], a2[4];
#pragma unroll
        for (int q = 0; q < 4; q++) {
            const int hh = 2 * ho + (q >> 1), ww = 2 * wo + (q & 1);
            const size_t idx = ((size_t)(n * H + hh) * W + ww) * C + c;
            a1[q] = (y[idx]          - m1) * s1 * gg + bc;
            a2[q] = (y[idx + inHalf] - m2) * s2 * gg + bc;
        }
        const float mx = fmaxf(fmaxf(a1[0], a1[1]), fmaxf(a1[2], a1[3]));
        const int cnt = (a1[0] == mx) + (a1[1] == mx) + (a1[2] == mx) + (a1[3] == mx);
        float ssum = 0.f, pmax = 0.f;
#pragma unroll
        for (int q = 0; q < 4; q++) {
            const float tq = (cnt == 1 && a1[q] == mx) ? a2[q] : 0.f;
            ssum += (tq > 0.f ? 1.f : 0.f) - (tq < 0.f ? 1.f : 0.f);
            pmax = fmaxf(pmax, fabsf(tq));
        }
        const float zfo = (ssum > 0.f ? 1.f : 0.f) - (ssum < 0.f ? 1.f : 0.f);
        const float x1 = mx > 0.f ? mx : 0.f;
        const float x2 = (x1 > 0.f) ? zfo * pmax : 0.f;
        out[i]       = (bf16)x1;
        out[i + tot] = (bf16)x2;
    }
}

// FC activation: rows<100 -> relu(y1); rows>=100 -> y2 * [y1>0].
__global__ void fc_act(const float* __restrict__ y, bf16* __restrict__ out, int Nn)
{
    const size_t tot = (size_t)200 * Nn;
    for (size_t i = blockIdx.x * (size_t)blockDim.x + threadIdx.x; i < tot;
         i += (size_t)gridDim.x * blockDim.x) {
        const int row = (int)(i / Nn);
        const float v = y[i];
        if (row < 100)
            out[i] = (bf16)(v > 0.f ? v : 0.f);
        else
            out[i] = (bf16)((y[i - (size_t)100 * Nn] > 0.f) ? v : 0.f);
    }
}

// f32 NCHW input -> bf16 [2*NHW][3] NHWC, both stream halves identical.
__global__ void cvt_input(const float* __restrict__ x, bf16* __restrict__ out,
                          int N, int C, int H, int W)
{
    const int HW = H * W;
    const size_t NHW = (size_t)N * HW;
    const size_t tot = 2 * NHW * C;
    for (size_t i = blockIdx.x * (size_t)blockDim.x + threadIdx.x; i < tot;
         i += (size_t)gridDim.x * blockDim.x) {
        const int c = (int)(i % C);
        size_t p = i / C;
        if (p >= NHW) p -= NHW;
        const size_t n = p / HW, hw = p - n * HW;
        out[i] = (bf16)x[(n * C + c) * HW + hw];
    }
}

// f32 OIHW conv weights -> bf16 [Cout][tap*Cin+ci]  (tap = ky*3+kx)
__global__ void cvt_convw(const float* __restrict__ w, bf16* __restrict__ out,
                          int Cout, int Cin)
{
    const int K = 9 * Cin;
    const size_t tot = (size_t)Cout * K;
    for (size_t i = blockIdx.x * (size_t)blockDim.x + threadIdx.x; i < tot;
         i += (size_t)gridDim.x * blockDim.x) {
        const int co = (int)(i / K);
        const int k  = (int)(i - (size_t)co * K);
        const int tap = k / Cin;
        const int ci  = k - tap * Cin;
        out[i] = (bf16)w[((size_t)co * Cin + ci) * 9 + tap];
    }
}

__global__ void cvt_f32_bf16(const float* __restrict__ a, bf16* __restrict__ o, size_t n)
{
    for (size_t i = blockIdx.x * (size_t)blockDim.x + threadIdx.x; i < n;
         i += (size_t)gridDim.x * blockDim.x)
        o[i] = (bf16)a[i];
}

// NHWC conv output [2*100*2*2][512] -> NCHW-flat FC input [200][2048]
__global__ void flatten_k(const bf16* __restrict__ in, bf16* __restrict__ out)
{
    const size_t tot = (size_t)200 * 2048;
    for (size_t i = blockIdx.x * (size_t)blockDim.x + threadIdx.x; i < tot;
         i += (size_t)gridDim.x * blockDim.x) {
        const int col = (int)(i % 2048);
        const int row = (int)(i / 2048);
        const int s = row / 100, n = row - s * 100;
        const int c = col >> 2, hw = col & 3;
        const int h = hw >> 1, w = hw & 1;
        out[i] = in[((size_t)(s * 400 + (n * 2 + h) * 2 + w)) * 512 + c];
    }
}

// ---------------------------------------------------------------------------
extern "C" void kernel_launch(void* const* d_in, const int* in_sizes, int n_in,
                              void* d_out, int out_size, void* d_ws, size_t ws_size,
                              hipStream_t stream)
{
    (void)in_sizes; (void)n_in; (void)out_size; (void)ws_size;

    const float* x = (const float*)d_in[0];
    const float *conv_w[13], *A_p[13], *bn_g[13], *bn_b[13];
    for (int i = 0; i < 13; i++) {
        conv_w[i] = (const float*)d_in[1 + i];
        A_p[i]    = (const float*)d_in[14 + i];
        bn_g[i]   = (const float*)d_in[27 + i];
        bn_b[i]   = (const float*)d_in[40 + i];
    }
    const float *ca_w[4], *Aq[4];
    for (int j = 0; j < 4; j++) {
        ca_w[j] = (const float*)d_in[53 + j];
        Aq[j]   = (const float*)d_in[57 + j];
    }

    static const int CHin[13]  = {3, 64, 64, 128, 128, 256, 256, 256, 512, 512, 512, 512, 512};
    static const int CHout[13] = {64, 64, 128, 128, 256, 256, 256, 512, 512, 512, 512, 512, 512};
    static const bool isPool[13] = {0, 0, 0, 1, 0, 0, 1, 0, 0, 1, 0, 0, 1};
    static const int FCI[4] = {2048, 1024, 512, 128};
    static const int FCO[4] = {1024, 512, 128, 10};

    size_t off = 0;
    auto carve = [&](size_t bytes) -> void* {
        void* p = (char*)d_ws + off;
        off += (bytes + 255) & ~(size_t)255;
        return p;
    };

    const int Nb = 100;
    const size_t maxActElems = (size_t)2 * Nb * 32 * 32 * 128;   // 26.2M elems
    bf16*  actA = (bf16*)carve(maxActElems * sizeof(bf16));
    bf16*  actB = (bf16*)carve(maxActElems * sizeof(bf16));
    float* raw  = (float*)carve(maxActElems * sizeof(float));
    float* mean = (float*)carve(2 * 512 * sizeof(float));
    float* rstd = (float*)carve(2 * 512 * sizeof(float));
    bf16* wconv[13];
    for (int i = 0; i < 13; i++)
        wconv[i] = (bf16*)carve((size_t)CHout[i] * CHin[i] * 9 * sizeof(bf16));
    bf16* wfc[4];
    for (int j = 0; j < 4; j++)
        wfc[j] = (bf16*)carve((size_t)FCO[j] * FCI[j] * sizeof(bf16));
    bf16*  fcA   = (bf16*)carve((size_t)200 * 2048 * sizeof(bf16));
    bf16*  fcB   = (bf16*)carve((size_t)200 * 2048 * sizeof(bf16));
    float* fcraw = (float*)carve((size_t)200 * 1024 * sizeof(float));

    // --- weight/input conversion + relayout (every call; deterministic) ---
    for (int i = 0; i < 13; i++) {
        size_t n = (size_t)CHout[i] * CHin[i] * 9;
        cvt_convw<<<(int)((n + 255) / 256), 256, 0, stream>>>(conv_w[i], wconv[i], CHout[i], CHin[i]);
    }
    for (int j = 0; j < 4; j++) {
        size_t n = (size_t)FCO[j] * FCI[j];
        cvt_f32_bf16<<<(int)((n + 255) / 256), 256, 0, stream>>>(ca_w[j], wfc[j], n);
    }
    {
        size_t n = (size_t)2 * Nb * 32 * 32 * 3;
        cvt_input<<<(int)((n + 255) / 256), 256, 0, stream>>>(x, actA, Nb, 3, 32, 32);
    }

    // --- conv stack ---
    int H = 32, W = 32;
    bf16* cur = actA;
    bf16* nxt = actB;
    for (int i = 0; i < 13; i++) {
        const int Cin = CHin[i], Cout = CHout[i];
        const int NHW = Nb * H * W;
        const int M   = 2 * NHW;
        dim3 g((M + 127) / 128, (Cout + 63) / 64);
        conv_wmma<<<g, TB, 0, stream>>>(cur, wconv[i], A_p[i], raw, Nb, H, W, Cin, Cout);
        bn_stats<<<2 * Cout, 256, 0, stream>>>(raw, NHW, Cout, mean, rstd);
        if (isPool[i]) {
            size_t tot = (size_t)Nb * (H / 2) * (W / 2) * Cout;
            bn_pool<<<(int)((tot + 255) / 256), 256, 0, stream>>>(raw, mean, rstd, bn_g[i], bn_b[i],
                                                                  nxt, Nb, H, W, Cout);
            H /= 2; W /= 2;
        } else {
            size_t half = (size_t)NHW * Cout;
            bn_act<<<(int)((half + 255) / 256), 256, 0, stream>>>(raw, mean, rstd, bn_g[i], bn_b[i],
                                                                  nxt, half, Cout);
        }
        bf16* tmp = cur; cur = nxt; nxt = tmp;
    }

    // --- flatten NHWC -> NCHW-flat for FC ---
    flatten_k<<<(200 * 2048 + 255) / 256, 256, 0, stream>>>(cur, fcA);

    // --- FC stack (M=200 = both streams; final layer writes f32 into d_out) ---
    bf16* fin  = fcA;
    bf16* fout = fcB;
    for (int j = 0; j < 4; j++) {
        const int K = FCI[j], Nn = FCO[j];
        float* ybuf = (j == 3) ? (float*)d_out : fcraw;
        dim3 g((200 + 127) / 128, (Nn + 63) / 64);
        fc_wmma<<<g, TB, 0, stream>>>(fin, wfc[j], Aq[j], ybuf, 200, 100, K, Nn);
        if (j < 3) {
            size_t tot = (size_t)200 * Nn;
            fc_act<<<(int)((tot + 255) / 256), 256, 0, stream>>>(fcraw, fout, Nn);
            bf16* tmp = fin; fin = fout; fout = tmp;
        }
    }
}
